// MultiHeadAttentionBlock_83150566850955
// MI455X (gfx1250) — matrix operand order repaired and checked by reading, not compiled, not run on hardware
//
#include <hip/hip_runtime.h>
#include <hip/hip_bf16.h>

typedef __bf16 bf16_t;
typedef __attribute__((ext_vector_type(16))) __bf16 v16bf;
typedef __attribute__((ext_vector_type(8)))  __bf16 v8bf;
typedef __attribute__((ext_vector_type(8)))  float  v8f;

union V16 { v16bf v; v8bf h[2]; };

#define NB 4
#define NQC 2048
#define NKC 2048
#define DC 512
#define HC 8

__device__ __forceinline__ float hmax16(float x) {
    x = fmaxf(x, __shfl_xor(x, 1, 32));
    x = fmaxf(x, __shfl_xor(x, 2, 32));
    x = fmaxf(x, __shfl_xor(x, 4, 32));
    x = fmaxf(x, __shfl_xor(x, 8, 32));
    return x;
}
__device__ __forceinline__ float hsum16(float x) {
    x += __shfl_xor(x, 1, 32);
    x += __shfl_xor(x, 2, 32);
    x += __shfl_xor(x, 4, 32);
    x += __shfl_xor(x, 8, 32);
    return x;
}

// ---------------- prep kernels ----------------
__global__ void k_cvt(const float* __restrict__ in, bf16_t* __restrict__ out, int n) {
    int i = blockIdx.x * blockDim.x + threadIdx.x;
    if (i < n) out[i] = (bf16_t)in[i];
}

// Wt[w][n][k] = W[w][k][n], converted to bf16 (4 weight matrices, 512x512)
__global__ void k_wtrans(const float* __restrict__ W0, const float* __restrict__ W1,
                         const float* __restrict__ W2, const float* __restrict__ W3,
                         bf16_t* __restrict__ Wt) {
    __shared__ float tile[32][33];
    const float* W = (blockIdx.z == 0) ? W0 : (blockIdx.z == 1) ? W1
                   : (blockIdx.z == 2) ? W2 : W3;
    int k0 = blockIdx.x * 32, n0 = blockIdx.y * 32;
    for (int i = threadIdx.y; i < 32; i += 8)
        tile[i][threadIdx.x] = W[(size_t)(k0 + i) * DC + n0 + threadIdx.x];
    __syncthreads();
    bf16_t* out = Wt + (size_t)blockIdx.z * DC * DC;
    for (int i = threadIdx.y; i < 32; i += 8)
        out[(size_t)(n0 + i) * DC + k0 + threadIdx.x] = (bf16_t)tile[threadIdx.x][i];
}

// ------------- double-buffered 16x64-per-wave GEMM core (K = 512, chunks of 32) -------------
// A: row-major bf16 [rows,512]; Wt: W transposed (Wt[n][k]) bf16. Result in C layout.
__device__ __forceinline__ void gemm_core(const bf16_t* __restrict__ A,
                                          const bf16_t* __restrict__ Wt,
                                          int rb, int nb, int hf, int col, v8f acc[4]) {
    const bf16_t* arow = A + (size_t)(rb + col) * DC + (hf ? 16 : 0);
    const bf16_t* wrow = Wt + (size_t)(nb + col) * DC + (hf ? 16 : 0);

    V16 a_cur, a_nxt;
    v16bf b_cur[4], b_nxt[4];
    a_cur.h[0] = *(const v8bf*)(arow);
    a_cur.h[1] = *(const v8bf*)(arow + 8);
#pragma unroll
    for (int nt = 0; nt < 4; ++nt)
        b_cur[nt] = *(const v16bf*)(wrow + (size_t)nt * 16 * DC);

#pragma unroll 2
    for (int k0 = 0; k0 < DC; k0 += 32) {
        int k1 = k0 + 32;
        if (k1 < DC) {  // prefetch next chunk's fragments while WMMAs run
            a_nxt.h[0] = *(const v8bf*)(arow + k1);
            a_nxt.h[1] = *(const v8bf*)(arow + k1 + 8);
#pragma unroll
            for (int nt = 0; nt < 4; ++nt)
                b_nxt[nt] = *(const v16bf*)(wrow + (size_t)nt * 16 * DC + k1);
        }
#pragma unroll
        for (int nt = 0; nt < 4; ++nt)
            acc[nt] = __builtin_amdgcn_wmma_f32_16x16x32_bf16(
                false, a_cur.v, false, b_cur[nt], (short)0, acc[nt], false, false);
        a_cur = a_nxt;
#pragma unroll
        for (int nt = 0; nt < 4; ++nt) b_cur[nt] = b_nxt[nt];
    }
}

// projection: out[row,n] = A @ W + bias; writes bf16 (and optionally f32 for the residual)
__global__ void k_proj(const bf16_t* __restrict__ A, const bf16_t* __restrict__ Wt,
                       const float* __restrict__ bias, bf16_t* __restrict__ outB,
                       float* __restrict__ outF) {
    int rb = blockIdx.x * 16, nb = blockIdx.y * 64;
    int lane = threadIdx.x, hf = lane >> 4, col = lane & 15;
    v8f acc[4] = {};
    gemm_core(A, Wt, rb, nb, hf, col, acc);
#pragma unroll
    for (int nt = 0; nt < 4; ++nt)
#pragma unroll
        for (int r = 0; r < 8; ++r) {
            int row = rb + r + hf * 8;
            int n   = nb + nt * 16 + col;
            float y = acc[nt][r] + bias[n];
            size_t idx = (size_t)row * DC + n;
            if (outB) outB[idx] = (bf16_t)y;
            if (outF) outF[idx] = y;
        }
}

// output projection: out = res + relu(A @ W + bias)
__global__ void k_oproj(const bf16_t* __restrict__ A, const bf16_t* __restrict__ Wt,
                        const float* __restrict__ bias, const float* __restrict__ res,
                        float* __restrict__ out) {
    int rb = blockIdx.x * 16, nb = blockIdx.y * 64;
    int lane = threadIdx.x, hf = lane >> 4, col = lane & 15;
    v8f acc[4] = {};
    gemm_core(A, Wt, rb, nb, hf, col, acc);
#pragma unroll
    for (int nt = 0; nt < 4; ++nt)
#pragma unroll
        for (int r = 0; r < 8; ++r) {
            int row = rb + r + hf * 8;
            int n   = nb + nt * 16 + col;
            float y = fmaxf(acc[nt][r] + bias[n], 0.f);
            size_t idx = (size_t)row * DC + n;
            out[idx] = res[idx] + y;
        }
}

// ---------------- flash attention ----------------
// grid: (NQ/128, H, B); block 256 (8 waves); wave handles 16 q rows, DS=64 head dim.
// Double-buffered: K fragments + mask bits in registers, V^T tile in LDS.
__global__ void k_attn(const bf16_t* __restrict__ qbf, const bf16_t* __restrict__ kbf,
                       const bf16_t* __restrict__ vbf, const float* __restrict__ qf32,
                       const int* __restrict__ mask, float* __restrict__ Opre) {
    __shared__ __align__(16) bf16_t VtL[2][64 * 40];    // double-buffered V^T: Vt[d][key]
    __shared__ __align__(16) bf16_t PL[8][16 * 40];     // per-wave P tile 16x32

    int b = blockIdx.z, h = blockIdx.y;
    int wv   = threadIdx.x >> 5;
    int lane = threadIdx.x & 31;
    int hf   = lane >> 4, col = lane & 15;
    int tid  = threadIdx.x;
    int qbase = blockIdx.x * 128 + wv * 16;
    const float scale = 0.04419417382415922f;           // 1/sqrt(512)

    // V staging coordinates: one coalesced 16B load per thread per chunk
    int kk_s = tid >> 3;          // key row 0..31
    int dd_s = (tid & 7) * 8;     // head-dim start 0,8,..,56

    // preload q A-fragments (two K-chunks of 32 covering head dim 64) — loop invariant
    V16 aq[2];
    {
        const bf16_t* qp = qbf + ((size_t)(b * NQC + qbase + col)) * DC + h * 64;
#pragma unroll
        for (int c = 0; c < 2; ++c) {
            const bf16_t* p = qp + c * 32 + (hf ? 16 : 0);
            aq[c].h[0] = *(const v8bf*)(p);
            aq[c].h[1] = *(const v8bf*)(p + 8);
        }
    }

    // K-fragment + mask-bit loader for one 32-key chunk
    auto load_chunk = [&](int kb, v16bf bk[2][2], unsigned mk[2]) {
#pragma unroll
        for (int t = 0; t < 2; ++t) {
#pragma unroll
            for (int c = 0; c < 2; ++c) {
                const bf16_t* kp = kbf +
                    ((size_t)(b * NKC + kb + t * 16 + col)) * DC + h * 64 + c * 32 + (hf ? 16 : 0);
                bk[t][c] = *(const v16bf*)kp;
            }
            unsigned bits = 0;
#pragma unroll
            for (int r = 0; r < 8; ++r) {
                int q   = qbase + r + hf * 8;
                int key = kb + t * 16 + col;
                bits |= (mask[((size_t)b * NQC + q) * NKC + key] != 0 ? 1u : 0u) << r;
            }
            mk[t] = bits;
        }
    };

    // prologue: stage V chunk 0 into VtL[0], preload K chunk 0
    {
        v8bf vst = *(const v8bf*)(vbf + ((size_t)(b * NKC + kk_s)) * DC + h * 64 + dd_s);
#pragma unroll
        for (int j = 0; j < 8; ++j) VtL[0][(dd_s + j) * 40 + kk_s] = vst[j];
    }
    v16bf bk_cur[2][2];
    unsigned mk_cur[2];
    load_chunk(0, bk_cur, mk_cur);
    __syncthreads();

    v8f acc[4] = {};
    float m[8], l[8];
#pragma unroll
    for (int r = 0; r < 8; ++r) { m[r] = -1e30f; l[r] = 0.f; }

    for (int kb = 0; kb < NKC; kb += 32) {
        int  buf      = (kb >> 5) & 1;
        bool has_next = (kb + 32) < NKC;

        // issue next chunk's global loads first so they overlap all compute below
        v16bf bk_nxt[2][2];
        unsigned mk_nxt[2];
        v8bf vst_n;
        if (has_next) {
            vst_n = *(const v8bf*)(vbf +
                ((size_t)(b * NKC + kb + 32 + kk_s)) * DC + h * 64 + dd_s);
            load_chunk(kb + 32, bk_nxt, mk_nxt);
        }

        // S = Q K^T : two 16x16 score tiles, K-dim 64 = 2 WMMA each
        v8f s[2];
#pragma unroll
        for (int t = 0; t < 2; ++t) {
            v8f sf = {};
            sf = __builtin_amdgcn_wmma_f32_16x16x32_bf16(
                false, aq[0].v, false, bk_cur[t][0], (short)0, sf, false, false);
            sf = __builtin_amdgcn_wmma_f32_16x16x32_bf16(
                false, aq[1].v, false, bk_cur[t][1], (short)0, sf, false, false);
            s[t] = sf;
        }

        // scale + mask (reference: scores/sqrt(D), then masked -> -1e9)
#pragma unroll
        for (int t = 0; t < 2; ++t)
#pragma unroll
            for (int r = 0; r < 8; ++r)
                s[t][r] = ((mk_cur[t] >> r) & 1) ? s[t][r] * scale : -1e9f;

        // online softmax (row stats reduced over 16-lane half-waves)
        float pr[2][8];
#pragma unroll
        for (int r = 0; r < 8; ++r) {
            float rowm = hmax16(fmaxf(s[0][r], s[1][r]));
            float mnew = fmaxf(m[r], rowm);
            float corr = __expf(m[r] - mnew);
            float p0 = __expf(s[0][r] - mnew);
            float p1 = __expf(s[1][r] - mnew);
            l[r] = l[r] * corr + hsum16(p0 + p1);
            m[r] = mnew;
#pragma unroll
            for (int nt = 0; nt < 4; ++nt) acc[nt][r] = acc[nt][r] * corr;
            pr[0][r] = p0; pr[1][r] = p1;
        }

        // P (C layout) -> LDS -> A layout for the next WMMA
#pragma unroll
        for (int t = 0; t < 2; ++t)
#pragma unroll
            for (int r = 0; r < 8; ++r)
                PL[wv][(r + hf * 8) * 40 + t * 16 + col] = (bf16_t)pr[t][r];
        __builtin_amdgcn_wave_barrier();
        __asm__ volatile("" ::: "memory");

        V16 ap;
        {
            const bf16_t* pp = &PL[wv][col * 40 + (hf ? 16 : 0)];
            ap.h[0] = *(const v8bf*)(pp);
            ap.h[1] = *(const v8bf*)(pp + 8);
        }
        // O += P @ V ; B fragment = Vt[d][key] contiguous in LDS
#pragma unroll
        for (int nt = 0; nt < 4; ++nt) {
            const bf16_t* vp = &VtL[buf][(nt * 16 + col) * 40 + (hf ? 16 : 0)];
            v16bf bvf = *(const v16bf*)vp;
            acc[nt] = __builtin_amdgcn_wmma_f32_16x16x32_bf16(
                false, ap.v, false, bvf, (short)0, acc[nt], false, false);
        }

        // stage next V chunk into the other LDS buffer; rotate register buffers
        if (has_next) {
#pragma unroll
            for (int j = 0; j < 8; ++j) VtL[buf ^ 1][(dd_s + j) * 40 + kk_s] = vst_n[j];
#pragma unroll
            for (int t = 0; t < 2; ++t) {
                bk_cur[t][0] = bk_nxt[t][0];
                bk_cur[t][1] = bk_nxt[t][1];
                mk_cur[t]    = mk_nxt[t];
            }
        }
        __syncthreads();
    }

    // finalize: Opre = q_proj(f32) + acc / l   (residual on projected Q)
    float inv[8];
#pragma unroll
    for (int r = 0; r < 8; ++r) inv[r] = 1.f / l[r];
#pragma unroll
    for (int nt = 0; nt < 4; ++nt)
#pragma unroll
        for (int r = 0; r < 8; ++r) {
            int q = qbase + r + hf * 8;
            int d = h * 64 + nt * 16 + col;
            size_t idx = ((size_t)(b * NQC + q)) * DC + d;
            Opre[idx] = qf32[idx] + acc[nt][r] * inv[r];
        }
}

// ---------------- layernorm over 512 features: one wave per row ----------------
__global__ void k_ln(const float* __restrict__ in, const float* __restrict__ g,
                     const float* __restrict__ bb, float* __restrict__ outF,
                     bf16_t* __restrict__ outB) {
    int row  = blockIdx.x;
    int lane = threadIdx.x;
    const float* x = in + (size_t)row * DC;
    float4 v[4];
    float s = 0.f, ss = 0.f;
#pragma unroll
    for (int j = 0; j < 4; ++j) {
        v[j] = ((const float4*)x)[j * 32 + lane];
        s  += v[j].x + v[j].y + v[j].z + v[j].w;
        ss += v[j].x * v[j].x + v[j].y * v[j].y + v[j].z * v[j].z + v[j].w * v[j].w;
    }
    for (int off = 1; off < 32; off <<= 1) {
        s  += __shfl_xor(s, off, 32);
        ss += __shfl_xor(ss, off, 32);
    }
    float mean = s * (1.f / 512.f);
    float var  = ss * (1.f / 512.f) - mean * mean;
    float rstd = rsqrtf(var + 1e-5f);
#pragma unroll
    for (int j = 0; j < 4; ++j) {
        int c = (j * 32 + lane) * 4;
        float4 o;
        o.x = (v[j].x - mean) * rstd * g[c + 0] + bb[c + 0];
        o.y = (v[j].y - mean) * rstd * g[c + 1] + bb[c + 1];
        o.z = (v[j].z - mean) * rstd * g[c + 2] + bb[c + 2];
        o.w = (v[j].w - mean) * rstd * g[c + 3] + bb[c + 3];
        ((float4*)(outF + (size_t)row * DC))[j * 32 + lane] = o;
        if (outB) {
            size_t base = (size_t)row * DC + c;
            outB[base + 0] = (bf16_t)o.x;
            outB[base + 1] = (bf16_t)o.y;
            outB[base + 2] = (bf16_t)o.z;
            outB[base + 3] = (bf16_t)o.w;
        }
    }
}

extern "C" void kernel_launch(void* const* d_in, const int* in_sizes, int n_in,
                              void* d_out, int out_size, void* d_ws, size_t ws_size,
                              hipStream_t stream) {
    const float* Q    = (const float*)d_in[0];
    const float* K    = (const float*)d_in[1];
    const int*   mask = (const int*)d_in[2];
    const float* Wq   = (const float*)d_in[3];
    const float* bq   = (const float*)d_in[4];
    const float* Wk   = (const float*)d_in[5];
    const float* bk   = (const float*)d_in[6];
    const float* Wv   = (const float*)d_in[7];
    const float* bv   = (const float*)d_in[8];
    const float* Wo   = (const float*)d_in[9];
    const float* bo   = (const float*)d_in[10];
    const float* g0   = (const float*)d_in[11];
    const float* b0   = (const float*)d_in[12];
    const float* g1   = (const float*)d_in[13];
    const float* b1   = (const float*)d_in[14];
    float* out = (float*)d_out;

    const size_t ND = (size_t)NB * NQC * DC;   // 4,194,304 elements

    char* p = (char*)d_ws;
    auto alloc = [&](size_t bytes) {
        char* r = p;
        p += (bytes + 255) & ~(size_t)255;
        return r;
    };
    bf16_t* Wt   = (bf16_t*)alloc((size_t)4 * DC * DC * sizeof(bf16_t));
    bf16_t* Qin  = (bf16_t*)alloc(ND * 2);
    bf16_t* Kin  = (bf16_t*)alloc(ND * 2);
    bf16_t* qbf  = (bf16_t*)alloc(ND * 2);
    bf16_t* kbf  = (bf16_t*)alloc(ND * 2);
    bf16_t* vbf  = (bf16_t*)alloc(ND * 2);
    float*  qf32 = (float*) alloc(ND * 4);
    float*  Opre = (float*) alloc(ND * 4);
    // aliases: these regions are dead once attention has run
    float*  OlnF = (float*) kbf;   // kbf+vbf region (16.8 MB) holds LN0 f32 output
    bf16_t* OlnB = qbf;            // LN0 bf16 output
    float*  O2   = (float*) Qin;   // Qin+Kin region holds pre-LN1 output

    const int rows = NB * NQC;                     // 8192
    dim3 gemm_grid(rows / 16, DC / 64);            // 512 x 8 waves

    // 1. precision prep
    k_cvt<<<dim3((unsigned)((ND + 255) / 256)), dim3(256), 0, stream>>>(Q, Qin, (int)ND);
    k_cvt<<<dim3((unsigned)((ND + 255) / 256)), dim3(256), 0, stream>>>(K, Kin, (int)ND);
    k_wtrans<<<dim3(16, 16, 4), dim3(32, 8), 0, stream>>>(Wq, Wk, Wv, Wo, Wt);

    // 2. projections (WMMA bf16, f32 accumulate)
    k_proj<<<gemm_grid, dim3(32), 0, stream>>>(Qin, Wt + 0 * DC * DC, bq, qbf, qf32);
    k_proj<<<gemm_grid, dim3(32), 0, stream>>>(Kin, Wt + 1 * DC * DC, bk, kbf, nullptr);
    k_proj<<<gemm_grid, dim3(32), 0, stream>>>(Kin, Wt + 2 * DC * DC, bv, vbf, nullptr);

    // 3. flash attention + residual on projected Q
    k_attn<<<dim3(NQC / 128, HC, NB), dim3(256), 0, stream>>>(qbf, kbf, vbf, qf32, mask, Opre);

    // 4. LN0 -> (f32, bf16)
    k_ln<<<dim3(rows), dim3(32), 0, stream>>>(Opre, g0, b0, OlnF, OlnB);

    // 5. O = O + relu(O @ Wo + bo)
    k_oproj<<<gemm_grid, dim3(32), 0, stream>>>(OlnB, Wt + 3 * DC * DC, bo, OlnF, O2);

    // 6. LN1 -> d_out
    k_ln<<<dim3(rows), dim3(32), 0, stream>>>(O2, g1, b1, out, nullptr);
}
